// ScaleDotProductAttention_27711128994159
// MI455X (gfx1250) — matrix operand order repaired and checked by reading, not compiled, not run on hardware
//
#include <hip/hip_runtime.h>

// ---------- types ----------
typedef __attribute__((ext_vector_type(16))) _Float16 v16h;
typedef __attribute__((ext_vector_type(8)))  float    v8f;
typedef __attribute__((ext_vector_type(4)))  float    f4;
typedef __attribute__((ext_vector_type(4)))  unsigned v4u;
typedef __attribute__((ext_vector_type(8)))  int      v8i;
typedef __attribute__((ext_vector_type(4)))  int      v4i;

// ---------- problem constants ----------
#define BB 4
#define HH 16
#define SS 2048
#define DD 64
#define NW 4            // waves per workgroup
#define QROWS (NW*16)   // 64 query rows per workgroup

// ---------- LDS layout ----------
#define RST   68                    // staged tile row stride in dwords (64 + 4 pad via TDM pad fields)
#define KBUF  (16*RST*4)            // 4352 B per K-tile buffer
#define VBUF  (32*RST*4)            // 8704 B per V-tile buffer
#define PBYTES (NW*64*1024)         // 64KB per wave of f16 p-fragments
#define KOFF  PBYTES
#define VOFF  (KOFF + 2*KBUF)
#define LOFF  (VOFF + 2*VBUF)
#define SMEM_BYTES (LOFF + NW*16*4) // 288512 B < 320KB WGP LDS

__device__ __forceinline__ v8f wmma16(v16h a, v16h b, v8f c) {
  // v_wmma_f32_16x16x32_f16  D = A*B + C
  return __builtin_amdgcn_wmma_f32_16x16x32_f16(false, a, false, b, (short)0, c, false, false);
}

// Issue one TDM 2D tile load: rows x 64 fp32, row-major, into LDS at lds_off.
// D# group0/group1 packed per CDNA5 ISA 8.3/8.4. pad: 4 dwords after every 64
// dwords -> LDS row stride 68 dwords (bank rotation + keeps 16B alignment).
__device__ __forceinline__ void tdm_load(const float* g, unsigned lds_off, int rows) {
  unsigned long long ga = (unsigned long long)(size_t)g;
  v4u g0;
  g0.x = 1u;                                    // count = 1 (valid descriptor)
  g0.y = lds_off;                               // lds_addr
  g0.z = (unsigned)(ga & 0xffffffffull);        // global_addr lo
  g0.w = (unsigned)(ga >> 32) | (2u << 30);     // global_addr hi | type=2 ("image")
  v8i g1;
  g1[0] = (int)((2u << 16)   // data_size = 4B
              | (1u << 20)   // pad_enable
              | (5u << 22)   // pad_interval: 64 dwords
              | (3u << 25)); // pad_amount: 4 dwords
  g1[1] = (int)(64u   << 16);                   // tensor_dim0 = 64 (lo16 in [31:16])
  g1[2] = (int)(2048u << 16);                   // dim0 hi = 0, tensor_dim1 = 2048
  g1[3] = (int)(64u   << 16);                   // dim1 hi = 0, tile_dim0 = 64
  g1[4] = rows;                                 // tile_dim1 = rows, tile_dim2 = 0
  g1[5] = 64;                                   // tensor_dim0_stride = 64
  g1[6] = 0;
  g1[7] = 0;
  v4i z4 = {0, 0, 0, 0};                        // groups 2/3 unused (2D tensor)
  v8i z8 = {0, 0, 0, 0, 0, 0, 0, 0};            // extra group (6-arg toolchain form)
  __builtin_amdgcn_tensor_load_to_lds(g0, g1, z4, z4, z8, 0);
}

// convert 16 consecutive fp32 (LDS) into a v16h fragment
__device__ __forceinline__ v16h cvt16(const float* p) {
  v16h r;
#pragma unroll
  for (int i = 0; i < 16; ++i) r[i] = (_Float16)p[i];
  return r;
}

__global__ void __launch_bounds__(NW*32)
attn_fused_kernel(const float* __restrict__ q, const float* __restrict__ k,
                  const float* __restrict__ v, const int* __restrict__ mask,
                  float* __restrict__ out, float* __restrict__ attn) {
  extern __shared__ char smem[];

  const int tid  = threadIdx.x;
  const int wid  = tid >> 5;
  const int lane = tid & 31;
  const int ln   = lane & 15;   // column / row-within-16 index
  const int hi   = lane >> 4;   // lane half

  const int wg   = blockIdx.x;
  const int qblk = wg & (SS / QROWS - 1);
  const int bh   = wg >> 5;                    // SS/QROWS == 32 blocks
  const size_t baseQK = (size_t)bh * SS * DD;
  const float* qb = q + baseQK;
  const float* kb = k + baseQK;
  const float* vb = v + baseQK;
  const int*   mb = mask + (size_t)bh * SS * SS;
  float* attnb    = attn + (size_t)bh * SS * SS;
  const int qrow0 = qblk * QROWS + wid * 16;   // this wave's first query row

  _Float16* pbuf = (_Float16*)(smem + wid * 65536);
  float*    lsum = (float*)(smem + LOFF);

  // ---- Q A-fragments (16x32 f16 each, contiguous k = 16*hi + s order) ----
  v16h aq0, aq1;
  {
    const float* qr = qb + (size_t)(qrow0 + ln) * DD;
#pragma unroll
    for (int i = 0; i < 8; ++i) {
      aq0[i]     = (_Float16)qr[hi * 16 + i];
      aq0[8 + i] = (_Float16)qr[hi * 16 + 8 + i];
      aq1[i]     = (_Float16)qr[32 + hi * 16 + i];
      aq1[8 + i] = (_Float16)qr[32 + hi * 16 + 8 + i];
    }
  }

  float ps[8];
#pragma unroll
  for (int i = 0; i < 8; ++i) ps[i] = 0.0f;

  const int* mtile = mb + (size_t)(qrow0 + ln) * SS;

  // =========== Loop 1: scores + exp, p -> LDS (A-fragment order) ===========
  if (tid == 0) tdm_load(kb, KOFF, 16);
  for (int kt = 0; kt < SS / 16; ++kt) {
    if (tid == 0) __builtin_amdgcn_s_wait_tensorcnt(0);
    __syncthreads();
    if (tid == 0 && kt < SS / 16 - 1)
      tdm_load(kb + (size_t)(kt + 1) * 16 * DD, KOFF + ((kt + 1) & 1) * KBUF, 16);

    if (kt + 2 + hi < SS / 16)
      __builtin_prefetch(mtile + (kt + 2 + hi) * 16, 0, 0);  // global_prefetch mask

    const float* kst = (const float*)(smem + KOFF + (kt & 1) * KBUF);
    // B fragments of K^T (32x16 f16): lane = k-row, elements = consecutive d
    v16h b0 = cvt16(kst + ln * RST + hi * 16);
    v16h b1 = cvt16(kst + ln * RST + 32 + hi * 16);

    v8f c = {};
    c = wmma16(aq0, b0, c);
    c = wmma16(aq1, b1, c);

    const int laneA = 16 * ((ln >> 3) & 1);
    const int ebase = (ln & 7) + 8 * (kt & 1);
    const int kcol  = kt * 16 + ln;
#pragma unroll
    for (int i = 0; i < 8; ++i) {
      const int m  = i + hi * 8;                       // row within 16-tile (C layout)
      const int mm = mb[(size_t)(qrow0 + m) * SS + kcol];
      float p = mm ? __expf(c[i] * 0.125f) : 0.0f;     // masked -> exact 0
      ps[i] += p;
      pbuf[(kt >> 1) * 512 + (m + laneA) * 16 + ebase] = (_Float16)p;
    }
  }

  // ---- row sums (softmax denominator), wave32 xor-reduction over 16 lanes ----
#pragma unroll
  for (int i = 0; i < 8; ++i) {
    float s = ps[i];
    s += __shfl_xor(s, 1);
    s += __shfl_xor(s, 2);
    s += __shfl_xor(s, 4);
    s += __shfl_xor(s, 8);
    ps[i] = s;
  }
  if ((lane & 15) == 0) {
#pragma unroll
    for (int i = 0; i < 8; ++i) lsum[wid * 16 + hi * 8 + i] = ps[i];
  }
  __syncthreads();
  const float invl = 1.0f / lsum[wid * 16 + ln];
  float invl8[8];
#pragma unroll
  for (int i = 0; i < 8; ++i) invl8[i] = 1.0f / lsum[wid * 16 + hi * 8 + i];

  // =========== Loop 2: out += p @ V, write normalized attn ===========
  v8f co[4] = {v8f{}, v8f{}, v8f{}, v8f{}};
  if (tid == 0) tdm_load(vb, VOFF, 32);
  for (int kt2 = 0; kt2 < SS / 32; ++kt2) {
    if (tid == 0) __builtin_amdgcn_s_wait_tensorcnt(0);
    __syncthreads();
    if (tid == 0 && kt2 < SS / 32 - 1)
      tdm_load(vb + (size_t)(kt2 + 1) * 32 * DD, VOFF + ((kt2 + 1) & 1) * VBUF, 32);

    const float* vst = (const float*)(smem + VOFF + (kt2 & 1) * VBUF);

    // A fragment of p: contiguous 32B per lane (stored pre-arranged in loop 1)
    v16h ap = *(const v16h*)(pbuf + kt2 * 512 + lane * 16);

#pragma unroll
    for (int dch = 0; dch < 4; ++dch) {
      v16h bv;  // B fragment of V (32 x 16): lane = d column, elements = k rows
#pragma unroll
      for (int e = 0; e < 16; ++e)
        bv[e] = (_Float16)vst[(hi * 8 + (e & 7) + (e >> 3) * 16) * RST + dch * 16 + ln];
      co[dch] = wmma16(ap, bv, co[dch]);
    }

    // normalized attention tile write (lane pair covers full 64B lines)
    float* arow = attnb + (size_t)(qrow0 + ln) * SS + kt2 * 32;
#define AWR(E0, COFF)                                                     \
    { f4 w; w.x = (float)ap[E0] * invl;   w.y = (float)ap[E0+1] * invl;   \
           w.z = (float)ap[E0+2] * invl; w.w = (float)ap[E0+3] * invl;    \
      *(f4*)(arow + (COFF)) = w; }
    AWR(0,  8 * hi)
    AWR(4,  8 * hi + 4)
    AWR(8,  16 + 8 * hi)
    AWR(12, 20 + 8 * hi)
#undef AWR
  }

  // ---- final out = (p @ V) / l ----
#pragma unroll
  for (int dch = 0; dch < 4; ++dch) {
#pragma unroll
    for (int i = 0; i < 8; ++i) {
      const int rowg = qrow0 + hi * 8 + i;
      out[((size_t)bh * SS + rowg) * DD + dch * 16 + ln] = co[dch][i] * invl8[i];
    }
  }
}

extern "C" void kernel_launch(void* const* d_in, const int* in_sizes, int n_in,
                              void* d_out, int out_size, void* d_ws, size_t ws_size,
                              hipStream_t stream) {
  const float* q    = (const float*)d_in[0];
  const float* k    = (const float*)d_in[1];
  const float* v    = (const float*)d_in[2];
  const int*   mask = (const int*)d_in[3];
  float* out  = (float*)d_out;
  float* attn = out + (size_t)BB * HH * SS * DD;  // tuple: (out, attn) flat

  dim3 grid(BB * HH * (SS / QROWS));  // 2048 workgroups
  dim3 block(NW * 32);                // 4 waves (wave32)
  attn_fused_kernel<<<grid, block, SMEM_BYTES, stream>>>(q, k, v, mask, out, attn);
}
